// WindowNonLocalDenoising_46420006535886
// MI455X (gfx1250) — 8-bit matrix operand order repaired and checked by reading, compiled, not run on hardware
//
#include <hip/hip_runtime.h>

typedef __attribute__((ext_vector_type(16))) int   v16i;
typedef __attribute__((ext_vector_type(4)))  int   v4i;
typedef __attribute__((ext_vector_type(8)))  float v8f;

__device__ __forceinline__ unsigned char sw_f32_e4m3(float x) {
  unsigned u = __float_as_uint(x);
  unsigned s = (u >> 31) << 7;
  float a = __uint_as_float(u & 0x7fffffffu);
  if (a != a) return (unsigned char)(s | 0x7f);
  if (a >= 448.f) return (unsigned char)(s | 0x7e);
  if (a < 0x1p-10f) return (unsigned char)s;
  int e = (int)((__float_as_uint(a) >> 23) & 0xff) - 127;
  if (e < -6) {
    unsigned q = (unsigned)(a * 512.f + 0.5f);
    if (q >= 8) return (unsigned char)(s | 0x08);
    return (unsigned char)(s | q);
  }
  float scaled = a * exp2f((float)(3 - e));
  unsigned q = (unsigned)(scaled + 0.5f);
  if (q >= 16) { q >>= 1; ++e; if (e > 8) return (unsigned char)(s | 0x7e); }
  return (unsigned char)(s | ((unsigned)(e + 7) << 3) | (q & 7u));
}

__device__ __forceinline__ unsigned char cvt1_fp8(float x) {
#if __has_builtin(__builtin_amdgcn_cvt_pk_fp8_f32)
  int p = __builtin_amdgcn_cvt_pk_fp8_f32(x, x, 0, false);
  return (unsigned char)(p & 0xff);
#else
  return sw_f32_e4m3(x);
#endif
}

__device__ __forceinline__ v16i load_a_f8(const unsigned char* rowbase, int kb) {
  v16i a;
  unsigned long long* p = (unsigned long long*)&a;
#pragma unroll
  for (int i = 0; i < 8; ++i)
    p[i] = *(const unsigned long long*)(rowbase + kb + 16 * i);
  return a;
}
__device__ __forceinline__ v16i load_b_f8(const unsigned char* colbase, int kb) {
  v16i b;
  unsigned long long* p = (unsigned long long*)&b;
#pragma unroll
  for (int i = 0; i < 8; ++i)
    p[i] = *(const unsigned long long*)(colbase + kb + 16 * i);
  return b;
}
#define WMMA_F8(A, B, C) \
  __builtin_amdgcn_wmma_f32_16x16x128_fp8_fp8((A), (B), (short)0, (C), false, false)

__global__ void wnl_cvt_weights(const float* __restrict__ wt,
                                const float* __restrict__ wp,
                                const float* __restrict__ wj,
                                unsigned char* __restrict__ w8) {
  int i = blockIdx.x * 256 + threadIdx.x;
  if (i < 32768)       w8[i] = cvt1_fp8(wt[i] * 64.f);
  else if (i < 65536)  w8[i] = cvt1_fp8(wp[i - 32768] * 64.f);
  else if (i < 131072) w8[i] = cvt1_fp8(wj[i - 65536] * 64.f);
}

#define NPAD   336
#define KPAD   384
#define XT_OFF 0
#define XN_OFF (XT_OFF + 336 * 256)
#define TH_OFF (XN_OFF + 256 * 384)
#define PH_OFF (TH_OFF + 336 * 128)
#define SG_OFF (PH_OFF + 336 * 128)
#define SMEM_BYTES (SG_OFF + 8 * 16 * 384)

__global__ void __launch_bounds__(256, 1)
wnl_main(const float* __restrict__ xg,
         const unsigned char* __restrict__ wth8,
         const unsigned char* __restrict__ wph8,
         const unsigned char* __restrict__ wpj8,
         float* __restrict__ out) {
  extern __shared__ unsigned char smem[];
  unsigned char* Xt  = smem + XT_OFF;
  unsigned char* Xn  = smem + XN_OFF;
  unsigned char* THs = smem + TH_OFF;
  unsigned char* PHs = smem + PH_OFF;

  const int tid  = threadIdx.x;
  const int lane = tid & 31;
  const int wv   = tid >> 5;
  const int lcol = lane & 15;
  const int lhi  = lane >> 4;
  const int kbA  = lhi ? 8 : 0;
  const int kbB  = lhi ? 8 : 0;

  const int wi = blockIdx.x;
  const int b  = wi / 100;
  const int nh = (wi / 10) % 10;
  const int nw = wi % 10;
  const int h0 = nh * 18, w0 = nw * 18;

  for (int c = 0; c < 256; ++c) {
    const float* xc = xg + (((size_t)b * 256 + c) * 180 + h0) * 180 + w0;
    for (int m = tid; m < KPAD; m += 256) {
      unsigned char q = 0;
      if (m < 324) q = cvt1_fp8(xc[(m / 18) * 180 + (m % 18)]);
      Xn[c * KPAD + m] = q;
      if (m < NPAD) Xt[m * 256 + c] = q;
    }
  }
  __syncthreads();

  for (int rt = wv; rt < 21; rt += 8) {
    const unsigned char* arow = &Xt[(rt * 16 + lcol) * 256];
    v16i a0 = load_a_f8(arow, kbA);
    v16i a1 = load_a_f8(arow + 128, kbA);
#pragma unroll 1
    for (int et = 0; et < 8; ++et) {
      const unsigned char* tc = wth8 + (et * 16 + lcol) * 256;
      const unsigned char* pc = wph8 + (et * 16 + lcol) * 256;
      v8f at = {};
      at = WMMA_F8(a0, load_b_f8(tc, kbB), at);
      at = WMMA_F8(a1, load_b_f8(tc + 128, kbB), at);
      v8f ap = {};
      ap = WMMA_F8(a0, load_b_f8(pc, kbB), ap);
      ap = WMMA_F8(a1, load_b_f8(pc + 128, kbB), ap);
#pragma unroll
      for (int r = 0; r < 8; ++r) {
        int row = rt * 16 + r + lhi * 8;
        int e   = et * 16 + lcol;
        THs[row * 128 + e] = cvt1_fp8(at[r]);
        PHs[row * 128 + e] = cvt1_fp8(ap[r]);
      }
    }
  }
  __syncthreads();

  const float ssc = 1.0f / (4096.0f * 11.3137084990f);
  unsigned char* stg = smem + SG_OFF + wv * (16 * KPAD);

  for (int rt = wv; rt < 21; rt += 8) {
    if (lcol < 12) {
#pragma unroll
      for (int r = 0; r < 8; ++r)
        *(unsigned*)&stg[(r + lhi * 8) * KPAD + 336 + lcol * 4] = 0u;
    }

    v16i aT = load_a_f8(&THs[(rt * 16 + lcol) * 128], kbA);
    const bool maskc = (lcol >= 4);

    float mx[8];
#pragma unroll
    for (int r = 0; r < 8; ++r) mx[r] = -1e30f;
#pragma unroll 2
    for (int mt = 0; mt < 21; ++mt) {
      v8f acc = {};
      acc = WMMA_F8(aT, load_b_f8(&PHs[(mt * 16 + lcol) * 128], kbB), acc);
      const bool dead = (mt == 20) && maskc;
#pragma unroll
      for (int r = 0; r < 8; ++r) {
        float s = dead ? -1e30f : acc[r] * ssc;
        mx[r] = fmaxf(mx[r], s);
      }
    }
#pragma unroll
    for (int r = 0; r < 8; ++r)
#pragma unroll
      for (int d = 1; d < 16; d <<= 1)
        mx[r] = fmaxf(mx[r], __shfl_xor(mx[r], d, 32));

    float sm[8];
#pragma unroll
    for (int r = 0; r < 8; ++r) sm[r] = 0.f;
#pragma unroll 2
    for (int mt = 0; mt < 21; ++mt) {
      v8f acc = {};
      acc = WMMA_F8(aT, load_b_f8(&PHs[(mt * 16 + lcol) * 128], kbB), acc);
      const bool dead = (mt == 20) && maskc;
#pragma unroll
      for (int r = 0; r < 8; ++r) {
        float e = dead ? 0.f : __expf(acc[r] * ssc - mx[r]);
        sm[r] += e;
        stg[(r + lhi * 8) * KPAD + mt * 16 + lcol] = cvt1_fp8(e);
      }
    }
#pragma unroll
    for (int r = 0; r < 8; ++r) {
#pragma unroll
      for (int d = 1; d < 16; d <<= 1) sm[r] += __shfl_xor(sm[r], d, 32);
      sm[r] = 1.0f / sm[r];
    }

    const unsigned char* prow = &stg[lcol * KPAD];
    v16i p0 = load_a_f8(prow, kbA);
    v16i p1 = load_a_f8(prow + 128, kbA);
    v16i p2 = load_a_f8(prow + 256, kbA);
#pragma unroll 2
    for (int ct = 0; ct < 16; ++ct) {
      const unsigned char* gcol = &Xn[(ct * 16 + lcol) * KPAD];
      v8f acc = {};
      acc = WMMA_F8(p0, load_b_f8(gcol, kbB), acc);
      acc = WMMA_F8(p1, load_b_f8(gcol + 128, kbB), acc);
      acc = WMMA_F8(p2, load_b_f8(gcol + 256, kbB), acc);
#pragma unroll
      for (int r = 0; r < 8; ++r)
        stg[(r + lhi * 8) * KPAD + ct * 16 + lcol] = cvt1_fp8(acc[r] * sm[r]);
    }

    const unsigned char* yrow = &stg[lcol * KPAD];
    v16i y0 = load_a_f8(yrow, kbA);
    v16i y1 = load_a_f8(yrow + 128, kbA);
    float* otile = (float*)stg;
    const int n_out = rt * 16 + lcol;
    const int hh = h0 + n_out / 18, ww = w0 + n_out % 18;
    const size_t obase = (size_t)b * 256 * 32400 + (size_t)hh * 180 + ww;
#pragma unroll 1
    for (int ot = 0; ot < 16; ++ot) {
      const unsigned char* jc = wpj8 + (ot * 16 + lcol) * 256;
      v8f acc = {};
      acc = WMMA_F8(y0, load_b_f8(jc, kbB), acc);
      acc = WMMA_F8(y1, load_b_f8(jc + 128, kbB), acc);
#pragma unroll
      for (int r = 0; r < 8; ++r)
        otile[(r + lhi * 8) * 16 + lcol] = acc[r];
      if (n_out < 324) {
#pragma unroll
        for (int g = 0; g < 8; ++g) {
          int ol = 2 * g + lhi;
          float v = otile[lcol * 16 + ol];
          size_t idx = obase + (size_t)(ot * 16 + ol) * 32400;
          out[idx] = xg[idx] + v * (1.0f / 64.0f);
        }
      }
    }
  }
}

extern "C" void kernel_launch(void* const* d_in, const int* in_sizes, int n_in,
                              void* d_out, int out_size, void* d_ws, size_t ws_size,
                              hipStream_t stream) {
  const float* x   = (const float*)d_in[0];
  const float* wth = (const float*)d_in[1];
  const float* wph = (const float*)d_in[2];
  const float* wpj = (const float*)d_in[3];
  float* out = (float*)d_out;
  unsigned char* w8 = (unsigned char*)d_ws;

  hipLaunchKernelGGL(wnl_cvt_weights, dim3(512), dim3(256), 0, stream,
                     wth, wph, wpj, w8);

  (void)hipFuncSetAttribute((const void*)wnl_main,
                            hipFuncAttributeMaxDynamicSharedMemorySize,
                            SMEM_BYTES);
  hipLaunchKernelGGL(wnl_main, dim3(400), dim3(256), SMEM_BYTES, stream,
                     x, w8, w8 + 32768, w8 + 65536, out);
}
